// GodSpeedCausalFFT_74758200754823
// MI455X (gfx1250) — matrix operand order repaired and checked by reading, not compiled, not run on hardware
//
#include <hip/hip_runtime.h>

typedef __attribute__((ext_vector_type(16))) _Float16 v16h;
typedef __attribute__((ext_vector_type(8)))  float    v8f;

#define B_   4
#define L_   4096
#define D_   1024
#define E3_  3072
#define KLEN 128
#define PL   (L_ + KLEN)   // 4224 padded conv length
#define M_   (B_ * L_)     // 16384

union FragA { v16h v; unsigned u[8]; };
union FragC { v8f  v; float    f[8]; };

// ---- gfx1250 async global->LDS copy (ASYNCcnt-tracked, no VGPR round-trip) ----
// vdst operand = LDS byte offset (low 32 bits of the generic shared pointer).
__device__ __forceinline__ void async_ld_b128(const void* lds_dst, const void* gsrc) {
    unsigned off = (unsigned)(size_t)lds_dst;
    asm volatile("global_load_async_to_lds_b128 %0, %1, off"
                 :: "v"(off), "v"(gsrc) : "memory");
}
__device__ __forceinline__ void wait_async0() {
    asm volatile("s_wait_asynccnt 0" ::: "memory");
}

// ---------------- f32 -> f16 convert ----------------
__global__ void k_cvt_f16(const float* __restrict__ s, _Float16* __restrict__ d, int n) {
    int i = blockIdx.x * blockDim.x + threadIdx.x;
    if (i < n) d[i] = (_Float16)s[i];
}

// ---------------- WMMA GEMM: C[m,n] = sum_k A[m,k] * Bw[n,k] ----------------
// A [M,K] f16 row-major, Bw [N,K] f16 row-major, C [M,N] f32.
// Block tile 128x256, 8 waves (2x4), wave tile 64x64 (4x4 wmma accumulators).
#define BM  128
#define BN  256
#define BK  32
#define LDT 40   // padded LDS row stride (halves): 80B rows -> conflict-free frags

__device__ __forceinline__ void stage_tiles(
    const _Float16* __restrict__ A, const _Float16* __restrict__ Bw,
    _Float16* As, _Float16* Bs, int m0, int n0, int K, int k0, int tid)
{
#pragma unroll
    for (int i = 0; i < 2; ++i) {          // A: 128 rows x 32 halves = 512 chunks
        int ch = tid + i * 256, row = ch >> 2, kc = (ch & 3) << 3;
        async_ld_b128(As + row * LDT + kc, A + (size_t)(m0 + row) * K + k0 + kc);
    }
#pragma unroll
    for (int i = 0; i < 4; ++i) {          // B: 256 rows x 32 halves = 1024 chunks
        int ch = tid + i * 256, row = ch >> 2, kc = (ch & 3) << 3;
        async_ld_b128(Bs + row * LDT + kc, Bw + (size_t)(n0 + row) * K + k0 + kc);
    }
}

__global__ __launch_bounds__(256) void k_gemm_f16f32(
    const _Float16* __restrict__ A, const _Float16* __restrict__ Bw,
    float* __restrict__ C, int M, int N, int K)
{
    __shared__ __align__(16) _Float16 As[2][BM * LDT];   // 2 x 10 KB
    __shared__ __align__(16) _Float16 Bs[2][BN * LDT];   // 2 x 20 KB

    const int tid   = threadIdx.x;
    const int lane  = tid & 31;
    const int wid   = tid >> 5;     // 0..7
    const int waveM = wid >> 2;     // 0..1 -> 64 rows
    const int waveN = wid & 3;      // 0..3 -> 64 cols
    const int r     = lane & 15;
    const int h     = lane >> 4;

    const int n0 = blockIdx.x * BN;
    const int m0 = blockIdx.y * BM;

    FragC acc[4][4];
#pragma unroll
    for (int mt = 0; mt < 4; ++mt)
#pragma unroll
        for (int nt = 0; nt < 4; ++nt)
#pragma unroll
            for (int j = 0; j < 8; ++j) acc[mt][nt].f[j] = 0.0f;

    stage_tiles(A, Bw, As[0], Bs[0], m0, n0, K, 0, tid);
    wait_async0();
    __syncthreads();

    const int nk = K / BK;
    for (int t = 0; t < nk; ++t) {
        const int cur = t & 1;
        if (t + 1 < nk)   // uniform branch: prefetch next tile into alternate buffer
            stage_tiles(A, Bw, As[cur ^ 1], Bs[cur ^ 1], m0, n0, K, (t + 1) * BK, tid);

        FragA af[4], bf[4];
        // A 16x32 f16 layout: lanes 0-15 K{0..7,16..23}, lanes 16-31 K{8..15,24..31}
#pragma unroll
        for (int mt = 0; mt < 4; ++mt) {
            const _Float16* sp = &As[cur][(waveM * 64 + mt * 16 + r) * LDT];
#pragma unroll
            for (int v = 0; v < 8; ++v) {
                int kp = (v < 4) ? (2 * v + 8 * h) : (16 + 2 * (v - 4) + 8 * h);
                af[mt].u[v] = *(const unsigned*)(sp + kp);
            }
        }
        // B 32x16 f16 layout: lanes 0-15 K{0..7,16..23}, lanes 16-31 K{8..15,24..31}, N=lane&15
#pragma unroll
        for (int nt = 0; nt < 4; ++nt) {
            const _Float16* sp = &Bs[cur][(waveN * 64 + nt * 16 + r) * LDT];
#pragma unroll
            for (int v = 0; v < 8; ++v)
                bf[nt].u[v] = *(const unsigned*)(sp + ((v < 4) ? (2 * v + 8 * h) : (16 + 2 * (v - 4) + 8 * h)));
        }
#pragma unroll
        for (int mt = 0; mt < 4; ++mt)
#pragma unroll
            for (int nt = 0; nt < 4; ++nt)
                acc[mt][nt].v = __builtin_amdgcn_wmma_f32_16x16x32_f16(
                    false, af[mt].v, false, bf[nt].v,
                    (short)0, acc[mt][nt].v, false, false);

        wait_async0();     // next tile fully in LDS (this wave)
        __syncthreads();   // ... and from all waves; frag reads already dscnt-drained
    }

    // C/D layout: lane L, VGPR j -> M = j + 8*(L>>4), N = L&15
#pragma unroll
    for (int mt = 0; mt < 4; ++mt)
#pragma unroll
        for (int nt = 0; nt < 4; ++nt)
#pragma unroll
            for (int j = 0; j < 8; ++j) {
                int m = m0 + waveM * 64 + mt * 16 + j + 8 * h;
                int n = n0 + waveN * 64 + nt * 16 + r;
                C[(size_t)m * N + n] = acc[mt][nt].f[j];
            }
}

// ---------------- src = k*v into causal-padded buffer [B][PL][D] ----------------
__global__ void k_src_prep(const float* __restrict__ qkv, float* __restrict__ srcp) {
    size_t i = (size_t)blockIdx.x * blockDim.x + threadIdx.x;  // B*PL*D exact
    int d       = (int)(i & (D_ - 1));
    size_t rest = i >> 10;
    int pl      = (int)(rest % PL);
    int b       = (int)(rest / PL);
    if (pl < KLEN) {
        srcp[i] = 0.0f;                       // zero halo => branch-free causal conv
    } else {
        size_t m = (size_t)b * L_ + (pl - KLEN);
        srcp[i] = qkv[m * E3_ + D_ + d] * qkv[m * E3_ + 2 * D_ + d];
    }
}

// ---------------- causal 128-tap conv + q-gating, f16 output ----------------
// Block: 64 d-channels x 64 l-values. Async-stage the 191-row src window into
// LDS; each thread produces 16 consecutive l-outputs via a sliding 16-deep tap
// shift register: one LDS read feeds 16 FMAs.
__global__ __launch_bounds__(256) void k_conv_gate(
    const float* __restrict__ srcp,   // [B][PL][D]
    const float* __restrict__ qkv,    // [M][3072] (q in cols 0..1023)
    const float* __restrict__ kern,   // [128]
    _Float16* __restrict__ outh)      // [M][D]
{
    __shared__ __align__(16) float Ls[191 * 64];
    __shared__ float kl[KLEN];

    const int tid = threadIdx.x;
    const int dl  = tid & 63;
    const int lw  = tid >> 6;         // 0..3 l-workers
    const int d0  = blockIdx.x * 64;
    const int l0  = blockIdx.y * 64;
    const int b   = blockIdx.z;

    if (tid < KLEN) kl[tid] = kern[tid];

    // async-stage padded rows [l0+1, l0+191] (= unpadded l0-127 .. l0+63)
    const float* gbase = srcp + ((size_t)b * PL + (size_t)(l0 + 1)) * D_ + d0;
    for (int ch = tid; ch < 191 * 16; ch += 256) {       // 16B chunks: 191 rows x 16
        int row = ch >> 4, c = (ch & 15) << 2;
        async_ld_b128(&Ls[row * 64 + c], gbase + (size_t)row * D_ + c);
    }
    wait_async0();
    __syncthreads();

    float acc[16], kw[16];
#pragma unroll
    for (int i = 0; i < 16; ++i) { acc[i] = 0.0f; kw[i] = 0.0f; }

    const int base = lw * 16;
#pragma unroll 1
    for (int j = 0; j < 143; ++j) {
        // kw[i] == kern[127 + i - j] (0 outside [0,127])
#pragma unroll
        for (int i = 15; i >= 1; --i) kw[i] = kw[i - 1];
        kw[0] = (j <= 127) ? kl[127 - j] : 0.0f;
        float s = Ls[(base + j) * 64 + dl];
#pragma unroll
        for (int i = 0; i < 16; ++i) acc[i] = fmaf(s, kw[i], acc[i]);
    }

    const int d = d0 + dl;
#pragma unroll
    for (int i = 0; i < 16; ++i) {
        size_t m = (size_t)b * L_ + (size_t)(l0 + base + i);
        float q  = qkv[m * E3_ + d];
        outh[m * D_ + d] = (_Float16)(q * acc[i]);
    }
}

extern "C" void kernel_launch(void* const* d_in, const int* in_sizes, int n_in,
                              void* d_out, int out_size, void* d_ws, size_t ws_size,
                              hipStream_t stream) {
    const float* x    = (const float*)d_in[0];   // [4,4096,1024]
    const float* Wqkv = (const float*)d_in[1];   // [3072,1024]
    const float* Wout = (const float*)d_in[2];   // [1024,1024]
    const float* kern = (const float*)d_in[3];   // [128]
    float* out = (float*)d_out;                  // [4,4096,1024] f32

    char* ws = (char*)d_ws;
    _Float16* xh   = (_Float16*)(ws);                        //  32 MB
    _Float16* wqh  = (_Float16*)(ws + ((size_t)32  << 20));  //   6 MB
    _Float16* woh  = (_Float16*)(ws + ((size_t)40  << 20));  //   2 MB
    float*    qkv  = (float*)   (ws + ((size_t)48  << 20));  // 192 MB
    float*    srcp = (float*)   (ws + ((size_t)240 << 20));  //  66 MB
    _Float16* outh = (_Float16*)(ws + ((size_t)312 << 20));  //  32 MB

    const int nx  = M_ * D_;    // 16777216
    const int nwq = E3_ * D_;   //  3145728
    const int nwo = D_ * D_;    //  1048576

    k_cvt_f16<<<nx  / 256, 256, 0, stream>>>(x,    xh,  nx);
    k_cvt_f16<<<nwq / 256, 256, 0, stream>>>(Wqkv, wqh, nwq);
    k_cvt_f16<<<nwo / 256, 256, 0, stream>>>(Wout, woh, nwo);

    // qkv = x @ Wqkv^T   [16384,3072]
    k_gemm_f16f32<<<dim3(E3_ / BN, M_ / BM), 256, 0, stream>>>(xh, wqh, qkv, M_, E3_, D_);
    // src = k*v into zero-padded causal buffer
    k_src_prep<<<(B_ * PL * D_) / 256, 256, 0, stream>>>(qkv, srcp);
    // res = causal conv(src, decay); out = q * res (f16)
    k_conv_gate<<<dim3(D_ / 64, L_ / 64, B_), 256, 0, stream>>>(srcp, qkv, kern, outh);
    // final = out @ Wout^T   [16384,1024]
    k_gemm_f16f32<<<dim3(D_ / BN, M_ / BM), 256, 0, stream>>>(outh, woh, out, M_, D_, D_);
}
